// up_pspHead_60851096650192
// MI455X (gfx1250) — compile-verified, matrix operand order checked by reading, not run on hardware
//
#include <hip/hip_runtime.h>

typedef __attribute__((ext_vector_type(16))) _Float16 v16h;
typedef __attribute__((ext_vector_type(8)))  float    v8f;

union FragU { v16h v; uint4 q[2]; };

static __device__ __forceinline__ uint32_t pkhalf2(float a, float b) {
    union { _Float16 h[2]; uint32_t u; } t;
    t.h[0] = (_Float16)a; t.h[1] = (_Float16)b;
    return t.u;
}

// ---------------------------------------------------------------------------
// One-time weight conversion f32 -> f16 (weights are reused by every pixel
// tile, so convert once instead of per-GEMM-step).
// ---------------------------------------------------------------------------
__global__ void cvt_f32_to_f16_kernel(const float* __restrict__ X,
                                      _Float16* __restrict__ Y, int n)
{
    const int i = blockIdx.x * blockDim.x + threadIdx.x;
    if (i < n) Y[i] = (_Float16)X[i];
}

// ---------------------------------------------------------------------------
// Implicit-GEMM convolution via WMMA (wave32).
//   D = W[CO x K] * im2col(X)[K x P],  K = C*KS*KS (multiple of 64),
//   P = N*H*W (multiple of 64).
// Macro-tile per workgroup: 128 (C_out) x 64 (pixels), 8 waves.
//  - A fragments: direct b128 global loads from pre-converted f16 weights
//    (two aligned 16B runs per lane, row index clamped for the CO=19 tail).
//  - B tile: im2col gather -> f16 pack -> LDS in WMMA fragment order, shared
//    by all 8 waves; two K-step buffers per barrier period.
//  - 8x v_wmma_f32_16x16x32_f16 between barriers (A reused across 4 B tiles).
// Fused BN scale/bias + optional ReLU epilogue.
// ---------------------------------------------------------------------------
template<int KS, int PAD>
__global__ __launch_bounds__(256)
void conv_wmma_kernel(const float* __restrict__ X, const _Float16* __restrict__ Wt,
                      const float* __restrict__ scale, const float* __restrict__ bias,
                      float* __restrict__ Y,
                      int N, int C, int H, int W, int CO, int relu)
{
    constexpr int KHW = KS * KS;
    const int HW = H * W;
    const int P  = N * HW;
    const int K  = C * KHW;

    const int tid  = threadIdx.x;
    const int lane = tid & 31;
    const int wv   = tid >> 5;      // wave id 0..7 -> C_out sub-tile
    const int p0   = blockIdx.x * 64;
    const int co0  = blockIdx.y * 128;

    // B tile, WMMA fragment order: [buf][subtile][lane][8 x u32 (16 halfs)]
    __shared__ __align__(16) uint32_t bLds[2][4 * 32 * 8];   // 2 x (32K x 64N) f16

    // ---- B writer role: pixel (p0+nloc), 8 consecutive K values, x2 bufs --
    const int nloc = tid & 63;
    const int g    = tid >> 6;            // 0..3 -> K sub-group
    const int kl0  = g * 8;
    const int pB   = p0 + nloc;
    const int nB_i = pB / HW;
    const int hwB  = pB - nB_i * HW;
    const int yB   = hwB / W;
    const int xB   = hwB - yB * W;
    const int nt   = nloc >> 4, nl = nloc & 15;
    const uint32_t bDst = (uint32_t)((nt * 32 + nl + (g & 1) * 16) * 8 + (g >> 1) * 4);

    // ---- A reader role: fragment row, two 16B runs per lane ---------------
    const int hf = lane >> 4;             // half-wave selector
    const int r  = lane & 15;
    const int mRow = min(co0 + wv * 16 + r, CO - 1);   // clamp; extra rows discarded
    const _Float16* aBase = Wt + (size_t)mRow * K + hf * 8;

    v8f acc[4] = {};

    for (int k0 = 0; k0 < K; k0 += 64) {
        __syncthreads();   // protect previous iteration's B fragment reads

        // ---------------- stage two B buffers (im2col gather + f16 pack) ---
#pragma unroll
        for (int s = 0; s < 2; ++s) {
            float xv[8];
#pragma unroll
            for (int e = 0; e < 8; ++e) {
                const int kk  = k0 + s * 32 + kl0 + e;
                const int ci  = kk / KHW;
                const int rem = kk - ci * KHW;
                const int ky  = rem / KS;
                const int kx  = rem - ky * KS;
                const int iy  = yB + ky - PAD;
                const int ix  = xB + kx - PAD;
                float v = 0.0f;
                if (iy >= 0 && iy < H && ix >= 0 && ix < W)
                    v = X[((size_t)(nB_i * C + ci) * H + iy) * W + ix];
                xv[e] = v;
            }
            *(uint4*)&bLds[s][bDst] =
                make_uint4(pkhalf2(xv[0], xv[1]), pkhalf2(xv[2], xv[3]),
                           pkhalf2(xv[4], xv[5]), pkhalf2(xv[6], xv[7]));
        }

        __syncthreads();

        // ---------------- consume: 2 x (1 A fragment x 4 B fragments) ------
#pragma unroll
        for (int s = 0; s < 2; ++s) {
            const _Float16* ap = aBase + k0 + s * 32;
            FragU fa;
            fa.q[0] = *(const uint4*)(ap);        // K = {hi*8 .. hi*8+7}
            fa.q[1] = *(const uint4*)(ap + 16);   // K = {16+hi*8 ..}
            if (s == 0 && k0 + 64 < K)            // prefetch next weight line
                __builtin_prefetch(ap + 64, 0, 1);
#pragma unroll
            for (int ntile = 0; ntile < 4; ++ntile) {
                FragU fb;
                const uint4* bp = (const uint4*)&bLds[s][(ntile * 32 + lane) * 8];
                fb.q[0] = bp[0]; fb.q[1] = bp[1];
                acc[ntile] = __builtin_amdgcn_wmma_f32_16x16x32_f16(
                    false, fa.v, false, fb.v, (short)0, acc[ntile], false, false);
            }
        }
    }

    // ---------------- epilogue: BN + ReLU, NCHW scatter --------------------
#pragma unroll
    for (int ntile = 0; ntile < 4; ++ntile) {
        const int p   = p0 + ntile * 16 + r;       // C/D column N = lane&15
        const int n_i = p / HW;
        const int hw  = p - n_i * HW;
        const int y   = hw / W;
        const int x   = hw - y * W;
#pragma unroll
        for (int v = 0; v < 8; ++v) {
            const int m = co0 + wv * 16 + hf * 8 + v;   // C/D row M = vgpr + 8*half
            if (m < CO && p < P) {
                float val = acc[ntile][v];
                const float s  = scale ? scale[m] : 1.0f;
                const float bb = bias  ? bias[m]  : 0.0f;
                val = val * s + bb;
                if (relu) val = fmaxf(val, 0.0f);
                Y[((size_t)(n_i * CO + m) * H + y) * W + x] = val;
            }
        }
    }
}

// ---------------------------------------------------------------------------
// Bilinear resize, align_corners=True. Output may target a channel range of a
// larger tensor (co_off / c_stride_out) so PSP features land inside `cat`.
// ---------------------------------------------------------------------------
__global__ void resize_bilinear_kernel(const float* __restrict__ X, float* __restrict__ Y,
                                       int N, int C, int IH, int IW, int OH, int OW,
                                       int co_off, int c_stride_out)
{
    const int idx = blockIdx.x * blockDim.x + threadIdx.x;
    const int total = N * C * OH * OW;
    if (idx >= total) return;
    int t = idx;
    const int x = t % OW; t /= OW;
    const int y = t % OH; t /= OH;
    const int c = t % C;
    const int n = t / C;

    const float sy = (OH > 1) ? (float)(IH - 1) / (float)(OH - 1) : 0.0f;
    const float sx = (OW > 1) ? (float)(IW - 1) / (float)(OW - 1) : 0.0f;
    const float fy = y * sy, fx = x * sx;
    const int y0 = (int)floorf(fy), x0 = (int)floorf(fx);
    const int y1 = min(y0 + 1, IH - 1), x1 = min(x0 + 1, IW - 1);
    const float wy = fy - (float)y0, wx = fx - (float)x0;

    const float* base = X + ((size_t)(n * C + c)) * IH * IW;
    const float v00 = base[y0 * IW + x0], v01 = base[y0 * IW + x1];
    const float v10 = base[y1 * IW + x0], v11 = base[y1 * IW + x1];
    const float v = (v00 * (1.0f - wy) + v10 * wy) * (1.0f - wx)
                  + (v01 * (1.0f - wy) + v11 * wy) * wx;
    Y[(((size_t)n * c_stride_out + co_off + c) * OH + y) * OW + x] = v;
}

// ---------------------------------------------------------------------------
// Local attention upsample core: softmax_j( q . k[j] ) applied to v[j], where
// j indexes a 3x3 dilation-2 (pad-2) neighborhood. One wave per output pixel;
// wave32 shfl_xor reductions over the KD channel dot-products.
// ---------------------------------------------------------------------------
__global__ __launch_bounds__(128)
void attention_kernel(const float* __restrict__ Q, const float* __restrict__ Kt,
                      const float* __restrict__ V, float* __restrict__ Y,
                      int N, int KD, int CO, int H, int W, int co_off, int co_stride)
{
    const int wave = threadIdx.x >> 5;
    const int lane = threadIdx.x & 31;
    const int pix  = blockIdx.x * 4 + wave;
    const int total = N * H * W;
    if (pix >= total) return;
    int t = pix;
    const int x = t % W; t /= W;
    const int y = t % H;
    const int n = t / H;

    float dot[9];
#pragma unroll
    for (int j = 0; j < 9; ++j) dot[j] = 0.0f;

    for (int c = lane; c < KD; c += 32) {
        const float q = Q[((n * KD + c) * H + y) * W + x];
#pragma unroll
        for (int j = 0; j < 9; ++j) {
            const int iy = y + (j / 3) * 2 - 2;
            const int ix = x + (j % 3) * 2 - 2;
            float kv = 0.0f;
            if (iy >= 0 && iy < H && ix >= 0 && ix < W)
                kv = Kt[((n * KD + c) * H + iy) * W + ix];
            dot[j] += q * kv;
        }
    }
#pragma unroll
    for (int j = 0; j < 9; ++j)
        for (int off = 16; off > 0; off >>= 1)
            dot[j] += __shfl_xor(dot[j], off, 32);   // all lanes end with full sum

    float mx = dot[0];
#pragma unroll
    for (int j = 1; j < 9; ++j) mx = fmaxf(mx, dot[j]);
    float att[9];
    float se = 0.0f;
#pragma unroll
    for (int j = 0; j < 9; ++j) { att[j] = expf(dot[j] - mx); se += att[j]; }
    const float inv = 1.0f / se;
#pragma unroll
    for (int j = 0; j < 9; ++j) att[j] *= inv;

    for (int c = lane; c < CO; c += 32) {
        float s = 0.0f;
#pragma unroll
        for (int j = 0; j < 9; ++j) {
            const int iy = y + (j / 3) * 2 - 2;
            const int ix = x + (j % 3) * 2 - 2;
            if (iy >= 0 && iy < H && ix >= 0 && ix < W)
                s += att[j] * V[((n * CO + c) * H + iy) * W + ix];
        }
        Y[(((size_t)n * co_stride + co_off + c) * H + y) * W + x] = s;
    }
}

// ---------------------------------------------------------------------------
// Adaptive average pool to S x S (torch bin boundaries). Reads a channel range
// of a larger tensor (c_off within Ctot channels).
// ---------------------------------------------------------------------------
__global__ void apool_kernel(const float* __restrict__ X, float* __restrict__ Y,
                             int N, int Ctot, int c_off, int C, int H, int W, int S)
{
    const int idx = blockIdx.x * blockDim.x + threadIdx.x;
    const int total = N * C * S * S;
    if (idx >= total) return;
    int t = idx;
    const int sx = t % S; t /= S;
    const int sy = t % S; t /= S;
    const int c  = t % C;
    const int n  = t / C;
    const int h0 = (sy * H) / S, h1 = ((sy + 1) * H + S - 1) / S;
    const int w0 = (sx * W) / S, w1 = ((sx + 1) * W + S - 1) / S;
    const float* base = X + (((size_t)n * Ctot + c_off + c)) * H * W;
    float s = 0.0f;
    for (int yy = h0; yy < h1; ++yy)
        for (int xx = w0; xx < w1; ++xx)
            s += base[yy * W + xx];
    Y[((n * C + c) * S + sy) * S + sx] = s / (float)((h1 - h0) * (w1 - w0));
}

// ---------------------------------------------------------------------------
// Tiny direct 1x1 conv + BN + ReLU for PSP branch (HW is 1..36; WMMA overkill).
// ---------------------------------------------------------------------------
__global__ void conv1x1_small_kernel(const float* __restrict__ X, const float* __restrict__ Wt,
                                     const float* __restrict__ sc, const float* __restrict__ bi,
                                     float* __restrict__ Y, int N, int C, int CO, int HW)
{
    const int idx = blockIdx.x * blockDim.x + threadIdx.x;
    const int total = N * CO * HW;
    if (idx >= total) return;
    int t = idx;
    const int p  = t % HW; t /= HW;
    const int co = t % CO;
    const int n  = t / CO;
    float s = 0.0f;
    for (int ci = 0; ci < C; ++ci)
        s += Wt[co * C + ci] * X[(n * C + ci) * HW + p];
    s = fmaxf(s * sc[co] + bi[co], 0.0f);
    Y[(n * CO + co) * HW + p] = s;
}

// ---------------------------------------------------------------------------
// Launcher
// ---------------------------------------------------------------------------
extern "C" void kernel_launch(void* const* d_in, const int* in_sizes, int n_in,
                              void* d_out, int out_size, void* d_ws, size_t ws_size,
                              hipStream_t stream)
{
    const float* c1  = (const float*)d_in[0];   // [4,256,128,128] (unused by forward)
    const float* c2  = (const float*)d_in[1];   // [4,512,64,64]
    const float* c3  = (const float*)d_in[2];   // [4,1024,32,32]
    const float* c4  = (const float*)d_in[3];   // [4,2048,16,16]
    const float* w5  = (const float*)d_in[4];
    const float* s5  = (const float*)d_in[5];
    const float* b5  = (const float*)d_in[6];
    const float* r4w = (const float*)d_in[7],  *r4s  = (const float*)d_in[8],  *r4b  = (const float*)d_in[9];
    const float* r4w2= (const float*)d_in[10], *r4s2 = (const float*)d_in[11], *r4b2 = (const float*)d_in[12];
    const float* r3w = (const float*)d_in[13], *r3s  = (const float*)d_in[14], *r3b  = (const float*)d_in[15];
    const float* r3w2= (const float*)d_in[16], *r3s2 = (const float*)d_in[17], *r3b2 = (const float*)d_in[18];
    const float* pw[4]  = { (const float*)d_in[19], (const float*)d_in[22], (const float*)d_in[25], (const float*)d_in[28] };
    const float* ps[4]  = { (const float*)d_in[20], (const float*)d_in[23], (const float*)d_in[26], (const float*)d_in[29] };
    const float* pb[4]  = { (const float*)d_in[21], (const float*)d_in[24], (const float*)d_in[27], (const float*)d_in[30] };
    const float* pcw = (const float*)d_in[31], *pcs = (const float*)d_in[32], *pcb = (const float*)d_in[33];
    const float* w6  = (const float*)d_in[34], *b6  = (const float*)d_in[35];
    (void)c1; (void)in_sizes; (void)n_in; (void)out_size; (void)ws_size;

    float* ws = (float*)d_ws;
    size_t o = 0;
    auto grab  = [&](size_t n) { float* p = ws + o; o += n; return p; };
    auto grabh = [&](size_t n) { _Float16* p = (_Float16*)(ws + o); o += (n + 1) / 2; return p; };

    float* out5  = grab((size_t)4 * 512 * 16 * 16);
    float* r4    = grab((size_t)4 * 2048 * 32 * 32);
    float* q4    = grab((size_t)4 * 128 * 32 * 32);
    float* k4    = grab((size_t)4 * 128 * 32 * 32);
    float* outr4 = grab((size_t)4 * 512 * 32 * 32);
    float* out4  = grab((size_t)4 * 512 * 32 * 32);
    float* r3    = grab((size_t)4 * 1024 * 64 * 64);
    float* q3    = grab((size_t)4 * 64 * 64 * 64);
    float* k3    = grab((size_t)4 * 64 * 64 * 64);
    float* outr3 = grab((size_t)4 * 512 * 64 * 64);
    float* cat   = grab((size_t)4 * 2560 * 64 * 64);
    float* pool  = grab((size_t)4 * 512 * 36);
    float* poolc = grab((size_t)4 * 512 * 36);
    float* outpc = grab((size_t)4 * 512 * 64 * 64);

    // f16 weight copies (converted once per launch; reused by every tile)
    _Float16* w5f   = grabh((size_t)512 * 2048 * 9);
    _Float16* r4wf  = grabh((size_t)128 * 1024);
    _Float16* r4w2f = grabh((size_t)128 * 2048);
    _Float16* r3wf  = grabh((size_t)64 * 512);
    _Float16* r3w2f = grabh((size_t)64 * 1024);
    _Float16* pcwf  = grabh((size_t)512 * 2560 * 9);
    _Float16* w6f   = grabh((size_t)19 * 512);

    auto cvt = [&](const float* src, _Float16* dst, int n) {
        cvt_f32_to_f16_kernel<<<(n + 255) / 256, 256, 0, stream>>>(src, dst, n);
    };
    cvt(w5,   w5f,   512 * 2048 * 9);
    cvt(r4w,  r4wf,  128 * 1024);
    cvt(r4w2, r4w2f, 128 * 2048);
    cvt(r3w,  r3wf,  64 * 512);
    cvt(r3w2, r3w2f, 64 * 1024);
    cvt(pcw,  pcwf,  512 * 2560 * 9);
    cvt(w6,   w6f,   19 * 512);

    auto conv3 = [&](const float* X, const _Float16* Wt, const float* sc, const float* bi,
                     float* Y, int N, int C, int H, int W, int CO, int relu) {
        dim3 g((N * H * W + 63) / 64, (CO + 127) / 128);
        conv_wmma_kernel<3, 1><<<g, 256, 0, stream>>>(X, Wt, sc, bi, Y, N, C, H, W, CO, relu);
    };
    auto conv1 = [&](const float* X, const _Float16* Wt, const float* sc, const float* bi,
                     float* Y, int N, int C, int H, int W, int CO, int relu) {
        dim3 g((N * H * W + 63) / 64, (CO + 127) / 128);
        conv_wmma_kernel<1, 0><<<g, 256, 0, stream>>>(X, Wt, sc, bi, Y, N, C, H, W, CO, relu);
    };
    auto resize = [&](const float* X, float* Y, int N, int C, int IH, int IW, int OH, int OW,
                      int co_off, int cstr) {
        int total = N * C * OH * OW;
        resize_bilinear_kernel<<<(total + 255) / 256, 256, 0, stream>>>(X, Y, N, C, IH, IW, OH, OW, co_off, cstr);
    };

    // ---- stage 1: out5 = relu(bn(conv3x3(c4, w5)))           [4,512,16,16]
    conv3(c4, w5f, s5, b5, out5, 4, 2048, 16, 16, 512, 1);

    // ---- stage 2: local_up(c3, c4, out5) -> out4             [4,512,32,32]
    conv1(c3, r4wf, r4s, r4b, q4, 4, 1024, 32, 32, 128, 0);            // q
    resize(c4, r4, 4, 2048, 16, 16, 32, 32, 0, 2048);                   // resize c4
    conv1(r4, r4w2f, r4s2, r4b2, k4, 4, 2048, 32, 32, 128, 0);          // k
    resize(out5, outr4, 4, 512, 16, 16, 32, 32, 0, 512);                // resize out
    attention_kernel<<<(4 * 32 * 32 + 3) / 4, 128, 0, stream>>>(q4, k4, outr4, out4,
                                                                4, 128, 512, 32, 32, 0, 512);

    // ---- stage 3: local_up(c2, c3, out4) -> cat[:,0:512]     [4,512,64,64]
    conv1(c2, r3wf, r3s, r3b, q3, 4, 512, 64, 64, 64, 0);
    resize(c3, r3, 4, 1024, 32, 32, 64, 64, 0, 1024);
    conv1(r3, r3w2f, r3s2, r3b2, k3, 4, 1024, 64, 64, 64, 0);
    resize(out4, outr3, 4, 512, 32, 32, 64, 64, 0, 512);
    attention_kernel<<<(4 * 64 * 64 + 3) / 4, 128, 0, stream>>>(q3, k3, outr3, cat,
                                                                4, 64, 512, 64, 64, 0, 2560);

    // ---- stage 4: PSP pyramid -> cat[:, 512*(i+1) : 512*(i+2)]
    const int ss[4] = {1, 2, 3, 6};
    for (int i = 0; i < 4; ++i) {
        const int s = ss[i];
        int ptot = 4 * 512 * s * s;
        apool_kernel<<<(ptot + 255) / 256, 256, 0, stream>>>(cat, pool, 4, 2560, 0, 512, 64, 64, s);
        conv1x1_small_kernel<<<(ptot + 255) / 256, 256, 0, stream>>>(pool, pw[i], ps[i], pb[i],
                                                                     poolc, 4, 512, 512, s * s);
        resize(poolc, cat, 4, 512, s, s, 64, 64, 512 * (i + 1), 2560);
    }

    // ---- stage 5: outpc = relu(bn(conv3x3(cat, pcw)))        [4,512,64,64]
    conv3(cat, pcwf, pcs, pcb, outpc, 4, 2560, 64, 64, 512, 1);

    // ---- stage 6: out = conv1x1(outpc, w6) + b6              [4,19,64,64]
    conv1(outpc, w6f, nullptr, b6, (float*)d_out, 4, 512, 64, 64, 19, 0);
}
